// Block_87179246174368
// MI455X (gfx1250) — matrix operand order repaired and checked by reading, not compiled, not run on hardware
//
#include <hip/hip_runtime.h>
#include <hip/hip_bf16.h>

// ---------------------------------------------------------------------------
// Types for CDNA5 WMMA (wave32): D(16x16 f32) = A(16x32 bf16) x B(32x16 bf16) + C
// ---------------------------------------------------------------------------
typedef __attribute__((ext_vector_type(16))) __bf16 v16bf;
typedef __attribute__((ext_vector_type(8)))  float  v8f;

union FragA16 { v16bf v; unsigned int u[8]; };
union U32H    { unsigned int u; __bf16 h[2]; };

__device__ __forceinline__ v8f wmma_bf16(const v16bf& a, const v16bf& b, const v8f& c) {
    // 8 args: (neg_a, A, neg_b, B, c_mod, C, reuse_a, reuse_b)
    return __builtin_amdgcn_wmma_f32_16x16x32_bf16(false, a, false, b, (short)0, c, false, false);
}

__device__ __forceinline__ float gelu_f(float v) {
    const float c = 0.7978845608028654f; // sqrt(2/pi)
    float u = c * (v + 0.044715f * v * v * v);
    return 0.5f * v * (1.0f + tanhf(u));
}

// ---------------------------------------------------------------------------
// CDNA5 async global->LDS staging (ASYNCcnt path), with generic-copy fallback.
// 16 bytes per lane per op; destination/source must be 16B aligned.
// Builtin signature (from hipcc diagnostic): param0 = v4i (vector_size 16) in
// AS1 (global), param1 = AS3 (LDS), then imm offset + imm cpol.
// ---------------------------------------------------------------------------
#if __has_builtin(__builtin_amdgcn_global_load_async_to_lds_b128)
#define HAS_ASYNC_LDS 1
#else
#define HAS_ASYNC_LDS 0
#endif

typedef int v4i_gcc __attribute__((vector_size(16)));

__device__ __forceinline__ void async_copy16(void* lds_dst, const void* gsrc) {
#if HAS_ASYNC_LDS
    typedef __attribute__((address_space(1))) v4i_gcc GV4;
    typedef __attribute__((address_space(3))) v4i_gcc LV4;
    __builtin_amdgcn_global_load_async_to_lds_b128(
        (GV4*)(unsigned long long)gsrc,
        (LV4*)(unsigned int)(unsigned long long)lds_dst,
        0, 0);
#else
    *(float4*)lds_dst = *(const float4*)gsrc;
#endif
}

__device__ __forceinline__ void wait_async() {
#if HAS_ASYNC_LDS
#if __has_builtin(__builtin_amdgcn_s_wait_asynccnt)
    __builtin_amdgcn_s_wait_asynccnt(0);
#else
    asm volatile("s_wait_asynccnt 0" ::: "memory");
#endif
#endif
}

// ---------------------------------------------------------------------------
// fp32 [K][N] -> bf16 transposed [N][K] (tiled via LDS, coalesced both sides)
// ---------------------------------------------------------------------------
__global__ __launch_bounds__(256)
void cvt_transpose_kernel(const float* __restrict__ src, __bf16* __restrict__ dst,
                          int K, int N) {
    __shared__ __bf16 tile[64][65];
    const int kb = blockIdx.x * 64;
    const int nb = blockIdx.y * 64;
    for (int i = threadIdx.x; i < 64 * 64; i += 256) {
        int r = i >> 6, c = i & 63;                      // r: k-offset, c: n-offset
        tile[c][r] = (__bf16)src[(size_t)(kb + r) * N + nb + c];
    }
    __syncthreads();
    for (int i = threadIdx.x; i < 64 * 64; i += 256) {
        int r = i >> 6, c = i & 63;                      // r: n-offset, c: k-offset
        dst[(size_t)(nb + r) * K + kb + c] = tile[r][c];
    }
}

// ---------------------------------------------------------------------------
// LayerNorm: one row (C=2048) per 256-thread block, fp32 stats, bf16 out
// ---------------------------------------------------------------------------
__global__ __launch_bounds__(256)
void layernorm_kernel(const float* __restrict__ x, const float* __restrict__ g,
                      const float* __restrict__ b, __bf16* __restrict__ out, int C) {
    const int row = blockIdx.x;
    const float* xr = x + (size_t)row * C;
    float sum = 0.f, sq = 0.f;
    for (int i = threadIdx.x; i < C; i += 256) {
        float v = xr[i];
        sum += v; sq += v * v;
    }
    #pragma unroll
    for (int off = 16; off >= 1; off >>= 1) {
        sum += __shfl_xor(sum, off, 32);
        sq  += __shfl_xor(sq,  off, 32);
    }
    __shared__ float ssum[8], ssq[8];
    if ((threadIdx.x & 31) == 0) { ssum[threadIdx.x >> 5] = sum; ssq[threadIdx.x >> 5] = sq; }
    __syncthreads();
    sum = 0.f; sq = 0.f;
    #pragma unroll
    for (int i = 0; i < 8; ++i) { sum += ssum[i]; sq += ssq[i]; }
    const float mu  = sum / (float)C;
    const float var = sq / (float)C - mu * mu;
    const float inv = rsqrtf(var + 1e-5f);
    __bf16* orow = out + (size_t)row * C;
    for (int i = threadIdx.x; i < C; i += 256)
        orow[i] = (__bf16)((xr[i] - mu) * inv * g[i] + b[i]);
}

// ---------------------------------------------------------------------------
// Tiled bf16 WMMA GEMM: C[M,N] = A[M,K] @ Bt[N,K]^T (+bias, epilogue variants)
// Block tile 128x256, BK=32, 256 threads = 8 waves (2 along M x 4 along N);
// wave tile 64x64 (4x4 WMMA). Double-buffered LDS, async global->LDS staging,
// one barrier per K step.
// ---------------------------------------------------------------------------
enum { EPI_BF16 = 0, EPI_GELU_BF16 = 1, EPI_RES_F32 = 2 };

template <int EPI>
__global__ __launch_bounds__(256)
void gemm_bf16_kernel(const __bf16* __restrict__ A,   // [M][K] row-major
                      const __bf16* __restrict__ Bt,  // [N][K] row-major (pre-transposed B)
                      const float* __restrict__ bias,
                      const float* __restrict__ residual,
                      __bf16* __restrict__ outb, float* __restrict__ outf,
                      int M, int N, int K) {
    constexpr int SA = 40;   // LDS row stride (elements): 80B, 16B-aligned chunks
    __shared__ __align__(16) __bf16 As[2][128 * SA];
    __shared__ __align__(16) __bf16 Bs[2][256 * SA];

    const int t    = threadIdx.x;
    const int lane = t & 31;
    const int w    = t >> 5;
    const int l    = lane & 15;
    const int hi   = lane >> 4;
    const int cm   = (w & 1) * 64;        // wave M offset (2 waves along M)
    const int cn   = (w >> 1) * 64;       // wave N offset (4 waves along N)
    const int m0   = blockIdx.y * 128;
    const int n0   = blockIdx.x * 256;

    // stage one 128x32 A tile and one 256x32 B tile (as 16B chunks per lane)
    auto stage = [&](int k0, int buf) {
        #pragma unroll
        for (int i = 0; i < 2; ++i) {                   // 512 chunks for A
            int linear = t + i * 256;
            int row = linear >> 2;                      // 4 chunks per 32-el row
            int c8  = (linear & 3) * 8;
            async_copy16(&As[buf][row * SA + c8],
                         A + (size_t)(m0 + row) * K + k0 + c8);
        }
        #pragma unroll
        for (int i = 0; i < 4; ++i) {                   // 1024 chunks for B
            int linear = t + i * 256;
            int row = linear >> 2;
            int c8  = (linear & 3) * 8;
            async_copy16(&Bs[buf][row * SA + c8],
                         Bt + (size_t)(n0 + row) * K + k0 + c8);
        }
    };

    v8f acc[4][4] = {};
    const int nk = K >> 5;

    stage(0, 0);
    wait_async();
    __syncthreads();

    for (int kt = 0; kt < nk; ++kt) {
        const int cur = kt & 1;
        if (kt + 1 < nk) stage((kt + 1) << 5, cur ^ 1);
        if (kt + 2 < nk) {  // CDNA5 global prefetch hint for the tile after next
            __builtin_prefetch(A  + (size_t)(m0 + (t >> 1)) * K + ((kt + 2) << 5), 0, 1);
            __builtin_prefetch(Bt + (size_t)(n0 + t) * K + ((kt + 2) << 5), 0, 1);
        }

        const __bf16* Ap = As[cur];
        const __bf16* Bp = Bs[cur];

        FragA16 a[4];
        #pragma unroll
        for (int mi = 0; mi < 4; ++mi) {
            int row = cm + 16 * mi + l;
            #pragma unroll
            for (int v = 0; v < 8; ++v) {
                int kk = ((v & 4) << 2) + hi * 8 + ((v & 3) << 1);   // A 16x32 layout
                a[mi].u[v] = *(const unsigned int*)(&Ap[row * SA + kk]);
            }
        }
        #pragma unroll
        for (int ni = 0; ni < 4; ++ni) {
            FragA16 bfr;
            int col = cn + 16 * ni + l;
            #pragma unroll
            for (int v = 0; v < 8; ++v) {
                int kk = ((v & 4) << 2) + hi * 8 + ((v & 3) << 1);   // B 32x16 layout
                bfr.u[v] = *(const unsigned int*)(&Bp[col * SA + kk]);
            }
            #pragma unroll
            for (int mi = 0; mi < 4; ++mi)
                acc[mi][ni] = wmma_bf16(a[mi].v, bfr.v, acc[mi][ni]);
        }

        wait_async();
        __syncthreads();
    }

    // --- epilogue ---
    #pragma unroll
    for (int mi = 0; mi < 4; ++mi) {
        #pragma unroll
        for (int ni = 0; ni < 4; ++ni) {
            int gcol = n0 + cn + 16 * ni + l;
            float bs = bias[gcol];
            #pragma unroll
            for (int r = 0; r < 8; ++r) {
                int grow = m0 + cm + 16 * mi + r + hi * 8;   // C-layout: M = r + hi*8
                float v = acc[mi][ni][r] + bs;
                if (EPI == EPI_GELU_BF16) v = gelu_f(v);
                if (EPI == EPI_RES_F32) {
                    size_t idx = (size_t)grow * N + gcol;
                    outf[idx] = v + residual[idx];
                } else {
                    outb[(size_t)grow * N + gcol] = (__bf16)v;
                }
            }
        }
    }
}

// ---------------------------------------------------------------------------
// Flash attention (causal): block = 128 q rows x one (b,h); 8 waves x 16 rows
// qkv bf16 [8192, 6144] (q | k | v, each h*128 wide); y bf16 [8192, 2048]
// ---------------------------------------------------------------------------
__global__ __launch_bounds__(256)
void flash_attn_kernel(const __bf16* __restrict__ qkv, __bf16* __restrict__ y) {
    constexpr int D = 128, TT = 2048, CQ = 6144;
    constexpr int SK = 136;  // Ks[64][128] row stride (272B: 16B-aligned chunks)
    constexpr int SV = 68;   // Vt[128][64] row stride (V transposed)
    constexpr int SP = 66;   // Ps[16][64] per-wave row stride
    __shared__ __align__(16) __bf16 Ks[64 * SK];
    __shared__ __align__(16) __bf16 Vt[128 * SV];
    __shared__ __align__(16) __bf16 Ps[8 * 16 * SP];

    const int t    = threadIdx.x;
    const int lane = t & 31;
    const int w    = t >> 5;
    const int l    = lane & 15;
    const int hi   = lane >> 4;
    const int b    = blockIdx.y >> 4;
    const int h    = blockIdx.y & 15;
    const int q0   = blockIdx.x * 128;
    const int qr0  = q0 + w * 16;              // this wave's first query row
    const size_t rowbase = (size_t)b * TT;

    // Q fragments (16 rows x 128 d) straight from global into A-layout regs
    FragA16 qa[4];
    {
        const __bf16* Qg = qkv + (rowbase + qr0 + l) * CQ + h * D;
        #pragma unroll
        for (int kc = 0; kc < 4; ++kc)
            #pragma unroll
            for (int v = 0; v < 8; ++v) {
                int kk = kc * 32 + ((v & 4) << 2) + hi * 8 + ((v & 3) << 1);
                qa[kc].u[v] = *(const unsigned int*)(Qg + kk);
            }
    }

    v8f o[8] = {};
    float mrow[8], lrow[8];
    #pragma unroll
    for (int r = 0; r < 8; ++r) { mrow[r] = -1e30f; lrow[r] = 0.f; }
    const float scale = 0.08838834764831845f;  // 1/sqrt(128)

    const int ktmax = q0 / 64 + 1;             // inclusive, covers causal range
    for (int kt = 0; kt <= ktmax; ++kt) {
        // --- stage K via async B128 (row-major = B-operand layout) ---
        const __bf16* Kg = qkv + (rowbase + kt * 64) * CQ + 2048 + h * D;
        #pragma unroll
        for (int i = 0; i < 4; ++i) {              // 1024 16B chunks
            int linear = t + i * 256;
            int n  = linear >> 4;                  // 16 chunks per 128-el row
            int c8 = (linear & 15) * 8;
            async_copy16(&Ks[n * SK + c8], Kg + (size_t)n * CQ + c8);
        }
        // --- stage V transposed (16-bit scatter) ---
        const __bf16* Vg = qkv + (rowbase + kt * 64) * CQ + 4096 + h * D;
        #pragma unroll
        for (int i = 0; i < 16; ++i) {             // 4096 dwords
            int linear = t + i * 256;
            int n  = linear >> 6;
            int d2 = linear & 63;
            U32H u; u.u = *(const unsigned int*)(Vg + (size_t)n * CQ + d2 * 2);
            Vt[(d2 * 2 + 0) * SV + n] = u.h[0];
            Vt[(d2 * 2 + 1) * SV + n] = u.h[1];
        }
        wait_async();
        __syncthreads();

        // --- S = Q @ K^T (16x64 per wave) ---
        v8f s[4] = {};
        #pragma unroll
        for (int kc = 0; kc < 4; ++kc) {
            #pragma unroll
            for (int ni = 0; ni < 4; ++ni) {
                FragA16 bf;
                int col = 16 * ni + l;
                #pragma unroll
                for (int v = 0; v < 8; ++v) {
                    int kk = kc * 32 + ((v & 4) << 2) + hi * 8 + ((v & 3) << 1);
                    bf.u[v] = *(const unsigned int*)(&Ks[col * SK + kk]);
                }
                s[ni] = wmma_bf16(qa[kc].v, bf.v, s[ni]);
            }
        }

        // --- scale + causal mask ---
        #pragma unroll
        for (int ni = 0; ni < 4; ++ni) {
            int kidx = kt * 64 + ni * 16 + l;
            #pragma unroll
            for (int r = 0; r < 8; ++r) {
                int qidx = qr0 + r + hi * 8;
                float v = s[ni][r] * scale;
                s[ni][r] = (kidx <= qidx) ? v : -1e30f;
            }
        }

        // --- online softmax (rows live in 16-lane groups of each half) ---
        #pragma unroll
        for (int r = 0; r < 8; ++r) {
            float rmax = s[0][r];
            #pragma unroll
            for (int ni = 1; ni < 4; ++ni) rmax = fmaxf(rmax, s[ni][r]);
            #pragma unroll
            for (int off = 8; off >= 1; off >>= 1)
                rmax = fmaxf(rmax, __shfl_xor(rmax, off, 32));
            float newm  = fmaxf(mrow[r], rmax);
            float alpha = __expf(mrow[r] - newm);
            float psum  = 0.f;
            #pragma unroll
            for (int ni = 0; ni < 4; ++ni) {
                float p = __expf(s[ni][r] - newm);
                s[ni][r] = p;
                psum += p;
            }
            #pragma unroll
            for (int off = 8; off >= 1; off >>= 1)
                psum += __shfl_xor(psum, off, 32);
            lrow[r] = lrow[r] * alpha + psum;
            mrow[r] = newm;
            #pragma unroll
            for (int dc = 0; dc < 8; ++dc) o[dc][r] *= alpha;
            // C-layout -> LDS so we can reload P in A-layout
            #pragma unroll
            for (int ni = 0; ni < 4; ++ni)
                Ps[w * 16 * SP + (r + hi * 8) * SP + ni * 16 + l] = (__bf16)s[ni][r];
        }
        asm volatile("s_wait_dscnt 0" ::: "memory");  // per-wave LDS RAW fence

        // --- O += P @ V ---
        #pragma unroll
        for (int kc2 = 0; kc2 < 2; ++kc2) {
            FragA16 pa;
            #pragma unroll
            for (int v = 0; v < 8; ++v) {
                int kk = kc2 * 32 + ((v & 4) << 2) + hi * 8 + ((v & 3) << 1);
                pa.u[v] = *(const unsigned int*)(&Ps[w * 16 * SP + l * SP + kk]);
            }
            #pragma unroll
            for (int dc = 0; dc < 8; ++dc) {
                FragA16 vb;
                int col = dc * 16 + l;
                #pragma unroll
                for (int v = 0; v < 8; ++v) {
                    int kk = kc2 * 32 + ((v & 4) << 2) + hi * 8 + ((v & 3) << 1);
                    vb.u[v] = *(const unsigned int*)(&Vt[col * SV + kk]);
                }
                o[dc] = wmma_bf16(pa.v, vb.v, o[dc]);
            }
        }
        __syncthreads();
    }

    // --- normalize + store y (bf16) ---
    #pragma unroll
    for (int dc = 0; dc < 8; ++dc) {
        int col = h * D + dc * 16 + l;
        #pragma unroll
        for (int r = 0; r < 8; ++r) {
            int grow = qr0 + r + hi * 8;
            float v = o[dc][r] / lrow[r];
            y[(rowbase + grow) * 2048 + col] = (__bf16)v;
        }
    }
}

// ---------------------------------------------------------------------------
// Launch sequence
// ---------------------------------------------------------------------------
extern "C" void kernel_launch(void* const* d_in, const int* in_sizes, int n_in,
                              void* d_out, int out_size, void* d_ws, size_t ws_size,
                              hipStream_t stream) {
    (void)in_sizes; (void)n_in; (void)out_size; (void)ws_size;
    const float* x        = (const float*)d_in[0];
    const float* ln1_g    = (const float*)d_in[1];
    const float* ln1_b    = (const float*)d_in[2];
    const float* w_attn   = (const float*)d_in[3];
    const float* b_attn   = (const float*)d_in[4];
    const float* w_proj   = (const float*)d_in[5];
    const float* b_proj   = (const float*)d_in[6];
    const float* ln2_g    = (const float*)d_in[7];
    const float* ln2_b    = (const float*)d_in[8];
    const float* w_fc     = (const float*)d_in[9];
    const float* b_fc     = (const float*)d_in[10];
    const float* w_fcp    = (const float*)d_in[11];
    const float* b_fcp    = (const float*)d_in[12];
    float* out = (float*)d_out;
    char*  ws  = (char*)d_ws;

    const int M = 8192;          // B*T
    const int C = 2048;

    // workspace layout (bytes); weights stored TRANSPOSED [N][K] as bf16.
    // M1 reuses the QKV+Y region after attention is consumed.
    const size_t OFF_WATTN   = 0;                       // [6144][2048] bf16 = 24MB
    const size_t OFF_WPROJ   = 25165824;                // [2048][2048] bf16 =  8MB
    const size_t OFF_WFC     = 33554432;                // [8192][2048] bf16 = 32MB
    const size_t OFF_WFCP    = 67108864;                // [2048][8192] bf16 = 32MB
    const size_t OFF_H       = 100663296;               // 8192*2048 bf16 = 32MB (ln1 & ln2 out)
    const size_t OFF_QKV     = 134217728;               // 8192*6144 bf16 = 96MB
    const size_t OFF_Y       = 234881024;               // 8192*2048 bf16 = 32MB
    const size_t OFF_M1      = OFF_QKV;                 // 8192*8192 bf16 = 128MB (reuse)
    const size_t OFF_X1      = 268435456;               // 8192*2048 f32  = 64MB

    __bf16* wattn_t = (__bf16*)(ws + OFF_WATTN);
    __bf16* wproj_t = (__bf16*)(ws + OFF_WPROJ);
    __bf16* wfc_t   = (__bf16*)(ws + OFF_WFC);
    __bf16* wfcp_t  = (__bf16*)(ws + OFF_WFCP);
    __bf16* h_bf    = (__bf16*)(ws + OFF_H);
    __bf16* qkv_bf  = (__bf16*)(ws + OFF_QKV);
    __bf16* y_bf    = (__bf16*)(ws + OFF_Y);
    __bf16* m1_bf   = (__bf16*)(ws + OFF_M1);
    float*  x1      = (float*) (ws + OFF_X1);

    // 1) weight conversion fp32 [K][N] -> bf16 transposed [N][K]
    cvt_transpose_kernel<<<dim3(32,  96), 256, 0, stream>>>(w_attn, wattn_t, C,     3 * C);
    cvt_transpose_kernel<<<dim3(32,  32), 256, 0, stream>>>(w_proj, wproj_t, C,     C);
    cvt_transpose_kernel<<<dim3(32, 128), 256, 0, stream>>>(w_fc,   wfc_t,   C,     4 * C);
    cvt_transpose_kernel<<<dim3(128, 32), 256, 0, stream>>>(w_fcp,  wfcp_t,  4 * C, C);

    // 2) ln1
    layernorm_kernel<<<M, 256, 0, stream>>>(x, ln1_g, ln1_b, h_bf, C);

    // 3) qkv = h @ w_attn + b_attn  -> bf16 [8192, 6144]
    gemm_bf16_kernel<EPI_BF16><<<dim3(24, 64), 256, 0, stream>>>(
        h_bf, wattn_t, b_attn, nullptr, qkv_bf, nullptr, M, 3 * C, C);

    // 4) causal flash attention -> y bf16 [8192, 2048]
    flash_attn_kernel<<<dim3(16, 64), 256, 0, stream>>>(qkv_bf, y_bf);

    // 5) x1 = x + y @ w_proj + b_proj  (f32)
    gemm_bf16_kernel<EPI_RES_F32><<<dim3(8, 64), 256, 0, stream>>>(
        y_bf, wproj_t, b_proj, x, nullptr, x1, M, C, C);

    // 6) ln2 (reuses h buffer)
    layernorm_kernel<<<M, 256, 0, stream>>>(x1, ln2_g, ln2_b, h_bf, C);

    // 7) m1 = gelu(h2 @ w_fc + b_fc)  -> bf16 [8192, 8192]
    gemm_bf16_kernel<EPI_GELU_BF16><<<dim3(32, 64), 256, 0, stream>>>(
        h_bf, wfc_t, b_fc, nullptr, m1_bf, nullptr, M, 4 * C, C);

    // 8) out = x1 + m1 @ w_fc_proj + b_fc_proj  (f32)
    gemm_bf16_kernel<EPI_RES_F32><<<dim3(8, 64), 256, 0, stream>>>(
        m1_bf, wfcp_t, b_fcp, x1, nullptr, out, M, C, 4 * C);
}
